// TabNet_7739531067493
// MI455X (gfx1250) — matrix operand order repaired and checked by reading, not compiled, not run on hardware
//
#include <hip/hip_runtime.h>

// ---------------------------------------------------------------------------
// TabNet inference on MI455X (gfx1250): bf16 WMMA GEMMs with fused epilogues.
// ---------------------------------------------------------------------------

#define EPSF 1e-5f
#define RELAXF 1.5f
#define SQRT_HALF 0.70710678118654752f

typedef __attribute__((ext_vector_type(16))) __bf16 v16bf;
typedef __attribute__((ext_vector_type(8)))  float  v8f;

union Frag { uint4 u[2]; v16bf v; };

enum { MODE_FT_FIRST = 0, MODE_FT_MID = 1, MODE_FT_LAST = 2, MODE_ATT = 3, MODE_FINAL = 4 };

__device__ __forceinline__ unsigned short f32_to_bf16(float f) {
  unsigned int u = __float_as_uint(f);
  u += 0x7FFFu + ((u >> 16) & 1u);   // round-to-nearest-even
  return (unsigned short)(u >> 16);
}

// Pack W (K x N, f32 row-major) into per-lane WMMA B-fragment order (bf16):
// dst[((nt*KT + kt)*32 + lane)*16 + e] = W[kt*32 + (lane>=16?16:0) + e][nt*16 + lane%16]
__global__ void pack_w_kernel(const float* __restrict__ W, unsigned short* __restrict__ dst,
                              int K, int N) {
  int i = blockIdx.x * 256 + threadIdx.x;
  if (i >= K * N) return;
  int KT   = K >> 5;
  int e    = i & 15;
  int lane = (i >> 4) & 31;
  int kt   = (i >> 9) % KT;
  int nt   = (i >> 9) / KT;
  int n = nt * 16 + (lane & 15);
  int k = kt * 32 + ((lane >> 4) << 4) + e;
  dst[i] = f32_to_bf16(W[(size_t)k * N + n]);
}

// feats = bn0(features); prior = 1; xin = bf16(feats); out_acc = 0
__global__ void init_kernel(const float* __restrict__ x, const float* __restrict__ g,
                            const float* __restrict__ b, const float* __restrict__ m,
                            const float* __restrict__ v, float* __restrict__ feats,
                            float* __restrict__ prior, unsigned short* __restrict__ xin,
                            float* __restrict__ outacc, int accN) {
  size_t i = (size_t)blockIdx.x * 256 + threadIdx.x;
  int n = (int)(i & 511);
  float f = g[n] * (x[i] - m[n]) * rsqrtf(v[n] + EPSF) + b[n];
  feats[i] = f;
  prior[i] = 1.0f;
  xin[i] = f32_to_bf16(f);
  if (i < (size_t)accN) outacc[i] = 0.0f;
}

// One wave32 per row of 512: sparsemax via tau-bisection, then
// mask -> d_out, prior *= (RELAX - mask), xin = bf16(mask * feats).
__global__ void sparsemax_kernel(const float* __restrict__ z, float* __restrict__ prior,
                                 const float* __restrict__ feats, float* __restrict__ mask_out,
                                 unsigned short* __restrict__ xin) {
  const int wave = threadIdx.x >> 5, lane = threadIdx.x & 31;
  const size_t row = (size_t)blockIdx.x * 8 + wave;
  const float* zr = z + row * 512;
  float zz[16];
  float mx = -3.0e38f;
#pragma unroll
  for (int i = 0; i < 16; ++i) { zz[i] = zr[lane + 32 * i]; mx = fmaxf(mx, zz[i]); }
#pragma unroll
  for (int off = 16; off; off >>= 1) mx = fmaxf(mx, __shfl_xor(mx, off, 32));
  // sum(relu(z - tau)) = 1 has its root in [max-1, max]
  float lo = mx - 1.0f, hi = mx;
  for (int it = 0; it < 30; ++it) {
    float mid = 0.5f * (lo + hi);
    float s = 0.0f;
#pragma unroll
    for (int i = 0; i < 16; ++i) s += fmaxf(zz[i] - mid, 0.0f);
#pragma unroll
    for (int off = 16; off; off >>= 1) s += __shfl_xor(s, off, 32);
    if (s > 1.0f) lo = mid; else hi = mid;
  }
  float tau = 0.5f * (lo + hi);
#pragma unroll
  for (int i = 0; i < 16; ++i) {
    size_t off = row * 512 + lane + 32 * i;
    float mk = fmaxf(zz[i] - tau, 0.0f);
    mask_out[off] = mk;
    prior[off] = prior[off] * (RELAXF - mk);
    xin[off] = f32_to_bf16(mk * feats[off]);
  }
}

// Tiled GEMM: C[32 x 128*NT] = Xbf[32 x K] @ Wp, bf16 WMMA, fp32 accumulate,
// with fused mode-dependent epilogue.
template <int NT>
__global__ __launch_bounds__(256)
void gemm_ft(const unsigned short* __restrict__ Xbf, int lda, int K,
             const unsigned short* __restrict__ Wp, int mode,
             const float* __restrict__ gn, const float* __restrict__ bn_,
             const float* __restrict__ mn, const float* __restrict__ vn,
             const float* __restrict__ prior,
             float* hout, unsigned short* hbf,
             float* acc, unsigned short* accbf, int write_accbf,
             const float* __restrict__ bias) {
  constexpr int NB = 128 * NT;
  __shared__ __align__(16) unsigned char smem[65536];
  unsigned short* As = (unsigned short*)smem;   // [32][K+8] bf16 during compute
  float* Cs = (float*)smem;                     // [32][NB] f32 during epilogue
  const int SK = K + 8;                         // +8 bf16 pad => bank-conflict-free frags
  const int t = threadIdx.x;
  const size_t rowbase = (size_t)blockIdx.x * 32;

  // Stage the 32 x K activation tile (uniform trip count -> EXEC stays all-1s).
  const int kvec = K >> 3;
  for (int q = t; q < 32 * kvec; q += 256) {
    int r = q / kvec, cb = q - r * kvec;
    uint4 val = *(const uint4*)(Xbf + (rowbase + r) * (size_t)lda + cb * 8);
    *(uint4*)(As + r * SK + cb * 8) = val;
  }
  __syncthreads();

  const int wave = t >> 5, lane = t & 31;
  const int lhalf = lane >> 4, l16 = lane & 15;
  const int KT = K >> 5;

  v8f accv[2][NT];
  const v8f vzero = {0.f, 0.f, 0.f, 0.f, 0.f, 0.f, 0.f, 0.f};
#pragma unroll
  for (int mt = 0; mt < 2; ++mt)
#pragma unroll
    for (int nt = 0; nt < NT; ++nt) accv[mt][nt] = vzero;

  for (int kt = 0; kt < KT; ++kt) {
    // A fragments (contiguous K to match packed B: lanes 0-15 K 0-15; 16-31 K 16-31)
    Frag afr[2];
    const int kb8 = lhalf << 4;
#pragma unroll
    for (int mt = 0; mt < 2; ++mt) {
      const unsigned short* ap = As + (mt * 16 + l16) * SK + kt * 32 + kb8;
      afr[mt].u[0] = *(const uint4*)ap;
      afr[mt].u[1] = *(const uint4*)(ap + 8);
    }
#pragma unroll
    for (int nt = 0; nt < NT; ++nt) {
      const int ntg = wave * NT + nt;
      const unsigned short* wp = Wp + (((size_t)ntg * KT + kt) * 32 + lane) * 16;
      Frag bfr;
      bfr.u[0] = *(const uint4*)wp;
      bfr.u[1] = *(const uint4*)(wp + 8);
      accv[0][nt] = __builtin_amdgcn_wmma_f32_16x16x32_bf16(
          false, afr[0].v, false, bfr.v, (short)0, accv[0][nt], false, false);
      accv[1][nt] = __builtin_amdgcn_wmma_f32_16x16x32_bf16(
          false, afr[1].v, false, bfr.v, (short)0, accv[1][nt], false, false);
    }
  }

  // Spill C tiles to LDS (f32 C/D layout: reg r -> M = r + 8*(lane>=16), N = lane%16)
  __syncthreads();
#pragma unroll
  for (int mt = 0; mt < 2; ++mt)
#pragma unroll
    for (int nt = 0; nt < NT; ++nt) {
      const int ntg = wave * NT + nt;
#pragma unroll
      for (int r = 0; r < 8; ++r)
        Cs[(mt * 16 + r + lhalf * 8) * NB + ntg * 16 + l16] = accv[mt][nt][r];
    }
  __syncthreads();

  if (mode <= MODE_FT_LAST) {                 // GLU feature block (NB == 512)
    for (int idx = t; idx < 32 * 256; idx += 256) {
      int n = idx & 255, m = idx >> 8;
      float c0 = Cs[m * NB + n];
      float c1 = Cs[m * NB + n + 256];
      float a  = gn[n]       * (c0 - mn[n])       * rsqrtf(vn[n]       + EPSF) + bn_[n];
      float gg = gn[n + 256] * (c1 - mn[n + 256]) * rsqrtf(vn[n + 256] + EPSF) + bn_[n + 256];
      float gl = a * (1.0f / (1.0f + __expf(-gg)));
      size_t off = (rowbase + m) * 256 + n;
      float hn = (mode == MODE_FT_FIRST) ? gl : hout[off] * SQRT_HALF + gl;
      hout[off] = hn;
      hbf[off]  = f32_to_bf16(hn);
      if (mode == MODE_FT_LAST && n < 128) {
        size_t aoff = (rowbase + m) * 128 + n;
        float an = acc[aoff] + fmaxf(hn, 0.0f);
        acc[aoff] = an;
        if (write_accbf) accbf[aoff] = f32_to_bf16(an);
      }
    }
  } else if (mode == MODE_ATT) {              // z = bn(att) * prior (NB == 512)
    for (int idx = t; idx < 32 * 512; idx += 256) {
      int n = idx & 511, m = idx >> 9;
      float z = gn[n] * (Cs[m * NB + n] - mn[n]) * rsqrtf(vn[n] + EPSF) + bn_[n];
      size_t off = (rowbase + m) * 512 + n;
      hout[off] = z * prior[off];
    }
  } else {                                    // MODE_FINAL: out = C + bias (NB == 128)
    for (int idx = t; idx < 32 * 128; idx += 256) {
      int n = idx & 127, m = idx >> 7;
      hout[(rowbase + m) * 128 + n] = Cs[m * NB + n] + bias[n];
    }
  }
}

// ---------------------------------------------------------------------------

extern "C" void kernel_launch(void* const* d_in, const int* in_sizes, int n_in,
                              void* d_out, int out_size, void* d_ws, size_t ws_size,
                              hipStream_t stream) {
  const float* features = (const float*)d_in[0];
  const float* bn0_g = (const float*)d_in[1];
  const float* bn0_b = (const float*)d_in[2];
  const float* bn0_m = (const float*)d_in[3];
  const float* bn0_v = (const float*)d_in[4];
  const float* Ws0   = (const float*)d_in[5];
  const float* Ws1   = (const float*)d_in[6];
  const float* Wu    = (const float*)d_in[7];
  const float* ft_g  = (const float*)d_in[8];
  const float* ft_b  = (const float*)d_in[9];
  const float* ft_m  = (const float*)d_in[10];
  const float* ft_v  = (const float*)d_in[11];
  const float* W_att = (const float*)d_in[12];
  const float* att_g = (const float*)d_in[13];
  const float* att_b = (const float*)d_in[14];
  const float* att_m = (const float*)d_in[15];
  const float* att_v = (const float*)d_in[16];
  const float* Wf    = (const float*)d_in[17];
  const float* bf    = (const float*)d_in[18];
  float* out = (float*)d_out;

  const int B = in_sizes[0] / 512;   // 32768
  const int MB = B / 32;

  char* p = (char*)d_ws;
  auto carve = [&](size_t bytes) -> void* {
    void* r = (void*)p;
    p += (bytes + 255) & ~(size_t)255;
    return r;
  };
  float* feats  = (float*)carve((size_t)B * 512 * 4);
  float* prior  = (float*)carve((size_t)B * 512 * 4);
  float* zbuf   = (float*)carve((size_t)B * 512 * 4);
  float* hbuf   = (float*)carve((size_t)B * 256 * 4);
  float* outacc = (float*)carve((size_t)B * 128 * 4);
  unsigned short* xin   = (unsigned short*)carve((size_t)B * 512 * 2);
  unsigned short* hbf16 = (unsigned short*)carve((size_t)B * 256 * 2);
  unsigned short* accbf = (unsigned short*)carve((size_t)B * 128 * 2);
  unsigned short* pWs0  = (unsigned short*)carve((size_t)512 * 512 * 2);
  unsigned short* pWs1  = (unsigned short*)carve((size_t)256 * 512 * 2);
  unsigned short* pWu   = (unsigned short*)carve((size_t)12 * 256 * 512 * 2);
  unsigned short* pWatt = (unsigned short*)carve((size_t)5 * 128 * 512 * 2);
  unsigned short* pWf   = (unsigned short*)carve((size_t)128 * 128 * 2);

  // Pre-pack all weights into WMMA B-fragment order (bf16, L2-resident).
  pack_w_kernel<<<(512 * 512) / 256, 256, 0, stream>>>(Ws0, pWs0, 512, 512);
  pack_w_kernel<<<(256 * 512) / 256, 256, 0, stream>>>(Ws1, pWs1, 256, 512);
  for (int i = 0; i < 12; ++i)
    pack_w_kernel<<<(256 * 512) / 256, 256, 0, stream>>>(
        Wu + (size_t)i * 256 * 512, pWu + (size_t)i * 256 * 512, 256, 512);
  for (int s = 0; s < 5; ++s)
    pack_w_kernel<<<(128 * 512) / 256, 256, 0, stream>>>(
        W_att + (size_t)s * 128 * 512, pWatt + (size_t)s * 128 * 512, 128, 512);
  pack_w_kernel<<<(128 * 128) / 256, 256, 0, stream>>>(Wf, pWf, 128, 128);

  init_kernel<<<(B * 512) / 256, 256, 0, stream>>>(
      features, bn0_g, bn0_b, bn0_m, bn0_v, feats, prior, xin, outacc, B * 128);

  auto ft_block = [&](const unsigned short* X, int lda, int K, const unsigned short* W,
                      int t, int blk, int mode, int wacc) {
    gemm_ft<4><<<MB, 256, 0, stream>>>(
        X, lda, K, W, mode,
        ft_g + ((size_t)t * 4 + blk) * 512, ft_b + ((size_t)t * 4 + blk) * 512,
        ft_m + ((size_t)t * 4 + blk) * 512, ft_v + ((size_t)t * 4 + blk) * 512,
        nullptr, hbuf, hbf16, outacc, accbf, wacc, nullptr);
  };
  auto transformer = [&](int t) {
    int lastmode = (t == 0) ? MODE_FT_MID : MODE_FT_LAST;  // step-0 x doesn't feed out_acc
    ft_block(xin,   512, 512, pWs0,                              t, 0, MODE_FT_FIRST, 0);
    ft_block(hbf16, 256, 256, pWs1,                              t, 1, MODE_FT_MID,   0);
    ft_block(hbf16, 256, 256, pWu + (size_t)(t * 2 + 0) * 256 * 512, t, 2, MODE_FT_MID, 0);
    ft_block(hbf16, 256, 256, pWu + (size_t)(t * 2 + 1) * 256 * 512, t, 3, lastmode,
             (t == 5) ? 1 : 0);
  };

  transformer(0);
  for (int s = 0; s < 5; ++s) {
    // att = bn(x[:,128:256] @ W_att[s]); z = att * prior
    gemm_ft<4><<<MB, 256, 0, stream>>>(
        hbf16 + 128, 256, 128, pWatt + (size_t)s * 128 * 512, MODE_ATT,
        att_g + (size_t)s * 512, att_b + (size_t)s * 512,
        att_m + (size_t)s * 512, att_v + (size_t)s * 512,
        prior, zbuf, nullptr, nullptr, nullptr, 0, nullptr);
    sparsemax_kernel<<<B / 8, 256, 0, stream>>>(
        zbuf, prior, feats, out + (size_t)B * 128 + (size_t)s * B * 512, xin);
    transformer(s + 1);
  }
  // final = out_acc @ Wf + bf
  gemm_ft<1><<<MB, 256, 0, stream>>>(
      accbf, 128, 128, pWf, MODE_FINAL,
      nullptr, nullptr, nullptr, nullptr, nullptr, out, nullptr, nullptr, nullptr, 0, bf);
}
